// Net_6maxFull_36661840838804
// MI455X (gfx1250) — matrix operand order repaired and checked by reading, not compiled, not run on hardware
//
#include <hip/hip_runtime.h>

typedef __attribute__((ext_vector_type(16))) _Float16 v16h;
typedef __attribute__((ext_vector_type(8)))  _Float16 v8h;
typedef __attribute__((ext_vector_type(8)))  float    v8f;

#define BATCH   131072
#define WAVES   8
#define NITER   2
#define NBLOCKS (BATCH / 16 / WAVES / NITER)   // 512

// tile index bases inside the LDS tile arena (each tile = 16x32 halfs = 512)
#define T_GEN 0     // 10 cells * 3 = 30
#define T_W1  30    // 4 chunks * 4 ntiles = 16
#define T_OPP 46    // 20 cells * 3 = 60
#define T_W1O 106   // 2 chunks * 2 ntiles = 4
#define T_W2  110   // 3 chunks * 1 ntile = 3
#define NTILES 113

struct Params {
  const float *x, *gen_h, *gen_c, *opp_h, *opp_c;
  const float *W_g0_ih, *W_g0_hh, *b_g0_ih, *b_g0_hh;
  const float *W_g_ih, *W_g_hh, *b_g_ih, *b_g_hh;
  const float *W_o0_ih, *W_o0_hh, *b_o0_ih, *b_o0_hh;
  const float *W_o_ih, *W_o_hh, *b_o_ih, *b_o_hh;
  const float *W1, *b1, *W1o, *b1o, *W2, *b2, *W3, *b3;
  float *out;
};

struct WaveScratch {
  _Float16 ain[16 * 32];      // A staging: [row][K] for the next cell
  float    gcm[48 * 16];      // gates, column-major [col][row]
  _Float16 cat_gen[16 * 128]; // concat of 10 gen h outputs (100, padded)
  _Float16 cat_opp[16 * 64];  // concat of 4 opp h outputs (40, padded)
  _Float16 cat2[16 * 96];     // [h1_gen(50), opp_avg(20)] padded
  float    h2cm[16 * 16];     // h2, column-major
};

struct SMem {
  _Float16 tiles[NTILES * 512];   // 115712 B
  float bias_cell[30 * 40];       // combined b_ih + b_hh per cell
  float b1[64], b1o[32], b2[16], w3[16];
  float b3; float pad0[3];
  WaveScratch ws[WAVES];          // 8 * 14336 B
};

#define WAVE_SYNC() asm volatile("s_wait_dscnt 0" ::: "memory")

__device__ __forceinline__ float fsig(float x) {
  return __builtin_amdgcn_rcpf(1.0f + __builtin_amdgcn_exp2f(-1.44269504089f * x));
}
__device__ __forceinline__ float ftanh(float x) {
  return 2.0f * __builtin_amdgcn_rcpf(1.0f + __builtin_amdgcn_exp2f(-2.88539008178f * x)) - 1.0f;
}
__device__ __forceinline__ v8f splat8(float v) { return (v8f){v, v, v, v, v, v, v, v}; }

// A-matrix 16x32 f16 layout: lane<16 row r holds K{0..7,16..23}; lane>=16 holds K{8..15,24..31}
__device__ __forceinline__ v16h load_A(const _Float16* base, int rowstride, int lane) {
  int r = lane & 15, hi = lane >> 4;
  const _Float16* p = base + r * rowstride + hi * 8;
  union { v16h v; v8h h[2]; } u;
  u.h[0] = *(const v8h*)p;
  u.h[1] = *(const v8h*)(p + 16);
  return u.v;
}
// B-matrix 32x16 f16: lane n<16 = col n, K{0..7,16..23}; lane n+16 = col n, K{8..15,24..31}
// tiles stored row-major [n][k] (n = output column = W row)
__device__ __forceinline__ v16h load_B(const _Float16* tile, int lane) {
  int n = lane & 15, koff = (lane >> 4) * 8;
  const _Float16* p = tile + n * 32 + koff;
  union { v16h v; v8h h[2]; } u;
  u.h[0] = *(const v8h*)p;
  u.h[1] = *(const v8h*)(p + 16);
  return u.v;
}
__device__ __forceinline__ v8f wmma16(v16h a, v16h b, v8f c) {
  return __builtin_amdgcn_wmma_f32_16x16x32_f16(false, a, false, b, (short)0, c, false, false);
}
// D tile (reg r, lane l) = elem(row = r + 8*(l>>4), col = base + (l&15)); store col-major
__device__ __forceinline__ void store_d_gcm(float* gcm, int colbase, int lane, v8f d) {
  int col = colbase + (lane & 15);
  float* p = gcm + col * 16 + (lane >> 4) * 8;
  *(float4*)p       = make_float4(d[0], d[1], d[2], d[3]);
  *(float4*)(p + 4) = make_float4(d[4], d[5], d[6], d[7]);
}

__device__ __forceinline__ void cell_gates(const _Float16* tiles3, const float* bias40,
                                           WaveScratch* w, int lane) {
  v16h a = load_A(w->ain, 32, lane);
#pragma unroll
  for (int t = 0; t < 3; ++t) {
    v16h b = load_B(tiles3 + t * 512, lane);
    int col = t * 16 + (lane & 15);
    float bv = (col < 40) ? bias40[col] : 0.0f;
    v8f d = wmma16(a, b, splat8(bv));
    store_d_gcm(w->gcm, t * 16, lane, d);
  }
  WAVE_SYNC();
}

// lane<16: (row=lane, k 0..4); lane>=16: (row=lane-16, k 5..9)
__device__ __forceinline__ void cell_elem(WaveScratch* w, const float* crow0, int lane,
                                          float hout[5]) {
  int row = lane & 15, kb = (lane >> 4) * 5;
#pragma unroll
  for (int j = 0; j < 5; ++j) {
    int k = kb + j;
    float iv = w->gcm[k * 16 + row];
    float fv = w->gcm[(10 + k) * 16 + row];
    float gv = w->gcm[(20 + k) * 16 + row];
    float ov = w->gcm[(30 + k) * 16 + row];
    float cc = crow0[row * 10 + k];
    float c2 = fsig(fv) * cc + fsig(iv) * ftanh(gv);
    hout[j] = fsig(ov) * ftanh(c2);
  }
}

__global__ __launch_bounds__(256) void net_kernel(Params P) {
  __shared__ SMem sm;
  const int tid  = threadIdx.x;
  const int lane = tid & 31;
  const int wid  = tid >> 5;

  // ---------------- pack weights into LDS f16 tiles ----------------
  // cell tiles: 30 cells x 3 tiles x 512
  for (int e = tid; e < 30 * 1536; e += 256) {
    int cell = e / 1536, rem = e % 1536, t = rem >> 9, q = rem & 511;
    int n = q >> 5, k = q & 31, g = t * 16 + n;
    const float *Wih, *Whh; int in;
    if (cell == 0)        { Wih = P.W_g0_ih;            in = 12; Whh = P.W_g0_hh; }
    else if (cell < 10)   { int j = cell - 1;
                            Wih = P.W_g_ih + j * 400;   in = 10; Whh = P.W_g_hh + j * 400; }
    else {
      int cc = cell - 10, p = cc >> 2, j = cc & 3;
      if (j == 0)         { Wih = P.W_o0_ih + p * 160;  in = 4;  Whh = P.W_o0_hh + p * 400; }
      else                { int o = (p * 3 + j - 1) * 400;
                            Wih = P.W_o_ih + o;         in = 10; Whh = P.W_o_hh + o; }
    }
    float v = 0.0f;
    if (g < 40) {
      if (k < in)           v = Wih[g * in + k];
      else if (k < in + 10) v = Whh[g * 10 + (k - in)];
    }
    int tb = (cell < 10) ? (T_GEN + cell * 3 + t) : (T_OPP + (cell - 10) * 3 + t);
    sm.tiles[tb * 512 + q] = (_Float16)v;
  }
  // W1 (50x100): 4 K-chunks x 4 N-tiles
  for (int e = tid; e < 8192; e += 256) {
    int c = e >> 11, t = (e >> 9) & 3, q = e & 511, n = q >> 5, k = q & 31;
    int ri = t * 16 + n, ci = c * 32 + k;
    float v = (ri < 50 && ci < 100) ? P.W1[ri * 100 + ci] : 0.0f;
    sm.tiles[(T_W1 + c * 4 + t) * 512 + q] = (_Float16)v;
  }
  // W1o (20x40): 2 x 2
  for (int e = tid; e < 2048; e += 256) {
    int c = e >> 10, t = (e >> 9) & 1, q = e & 511, n = q >> 5, k = q & 31;
    int ri = t * 16 + n, ci = c * 32 + k;
    float v = (ri < 20 && ci < 40) ? P.W1o[ri * 40 + ci] : 0.0f;
    sm.tiles[(T_W1O + c * 2 + t) * 512 + q] = (_Float16)v;
  }
  // W2 (10x70): 3 x 1
  for (int e = tid; e < 1536; e += 256) {
    int c = e >> 9, q = e & 511, n = q >> 5, k = q & 31;
    int ci = c * 32 + k;
    float v = (n < 10 && ci < 70) ? P.W2[n * 70 + ci] : 0.0f;
    sm.tiles[(T_W2 + c) * 512 + q] = (_Float16)v;
  }
  // combined biases
  for (int e = tid; e < 1200; e += 256) {
    int cell = e / 40, g = e % 40; float v;
    if (cell == 0)      v = P.b_g0_ih[g] + P.b_g0_hh[g];
    else if (cell < 10) { int j = cell - 1; v = P.b_g_ih[j * 40 + g] + P.b_g_hh[j * 40 + g]; }
    else {
      int cc = cell - 10, p = cc >> 2, j = cc & 3;
      if (j == 0) v = P.b_o0_ih[p * 40 + g] + P.b_o0_hh[p * 40 + g];
      else        { int o = (p * 3 + j - 1) * 40 + g; v = P.b_o_ih[o] + P.b_o_hh[o]; }
    }
    sm.bias_cell[e] = v;
  }
  if (tid < 64) sm.b1[tid]  = (tid < 50) ? P.b1[tid]  : 0.0f;
  if (tid < 32) sm.b1o[tid] = (tid < 20) ? P.b1o[tid] : 0.0f;
  if (tid < 16) { sm.b2[tid] = (tid < 10) ? P.b2[tid] : 0.0f;
                  sm.w3[tid] = (tid < 10) ? P.W3[tid] : 0.0f; }
  if (tid == 0) sm.b3 = P.b3[0];
  __syncthreads();

  WaveScratch* w = &sm.ws[wid];
  { // zero per-wave staging (covers all padding columns)
    v8h z = {};
    for (int i = lane; i < 64;  i += 32) *(v8h*)(w->ain     + i * 8) = z;
    for (int i = lane; i < 256; i += 32) *(v8h*)(w->cat_gen + i * 8) = z;
    for (int i = lane; i < 128; i += 32) *(v8h*)(w->cat_opp + i * 8) = z;
    for (int i = lane; i < 192; i += 32) *(v8h*)(w->cat2    + i * 8) = z;
  }
  WAVE_SYNC();

  const int row = lane & 15;
  const int kb  = (lane >> 4) * 5;

  for (int it = 0; it < NITER; ++it) {
    const int g0 = ((blockIdx.x * WAVES + wid) * NITER + it) * 16;
    float hout[5];

    // ---- generator cell0 staging: x[:,0:12] -> cols 0-11, gen_h[0] -> cols 12-21 ----
    {
      const float* xr = P.x + (size_t)(g0 + row) * 37;
      int cb = (lane >> 4) * 6;
#pragma unroll
      for (int j = 0; j < 6; ++j) w->ain[row * 32 + cb + j] = (_Float16)xr[cb + j];
      const float* hs = P.gen_h + (size_t)g0 * 10;
#pragma unroll
      for (int j = 0; j < 5; ++j)
        w->ain[row * 32 + 12 + kb + j] = (_Float16)hs[row * 10 + kb + j];
    }
    WAVE_SYNC();

    // ---- generator stack: 10 cells ----
    for (int j = 0; j < 10; ++j) {
      cell_gates(sm.tiles + (size_t)(T_GEN + j * 3) * 512, sm.bias_cell + j * 40, w, lane);
      cell_elem(w, P.gen_c + ((size_t)j * BATCH + g0) * 10, lane, hout);
      if (j < 9) {
        const float* hs = P.gen_h + ((size_t)(j + 1) * BATCH + g0) * 10;
#pragma unroll
        for (int q = 0; q < 5; ++q) {
          w->ain[row * 32 + kb + q]      = (_Float16)hout[q];
          w->ain[row * 32 + 10 + kb + q] = (_Float16)hs[row * 10 + kb + q];
        }
        if (j == 0) w->ain[row * 32 + 20 + (lane >> 4)] = (_Float16)0.0f; // clear cell0 residue
      }
#pragma unroll
      for (int q = 0; q < 5; ++q)
        w->cat_gen[row * 128 + 10 * j + kb + q] = (_Float16)hout[q];
      WAVE_SYNC();
    }

    // ---- h1_gen = tanh(cat_gen(100) @ W1^T + b1) : 4 K-chunks x 4 N-tiles ----
    v8f D1[4];
#pragma unroll
    for (int c = 0; c < 4; ++c) {
      v16h a = load_A(w->cat_gen + c * 32, 128, lane);
#pragma unroll
      for (int t = 0; t < 4; ++t) {
        v16h b = load_B(sm.tiles + (size_t)(T_W1 + c * 4 + t) * 512, lane);
        v8f cin = (c == 0) ? splat8(sm.b1[t * 16 + (lane & 15)]) : D1[t];
        D1[t] = wmma16(a, b, cin);
      }
    }
#pragma unroll
    for (int t = 0; t < 4; ++t) {
      int col = t * 16 + (lane & 15);
      if (col < 50) {
        int r0 = (lane >> 4) * 8;
#pragma unroll
        for (int r = 0; r < 8; ++r)
          w->cat2[(r0 + r) * 96 + col] = (_Float16)ftanh(D1[t][r]);
      }
    }

    // ---- opponent branches ----
    v8f oacc0 = splat8(0.0f), oacc1 = splat8(0.0f);
    for (int p = 0; p < 5; ++p) {
      { // cell0 staging: zero cols14-19, xo -> cols 0-3, opp_h[p][0] -> cols 4-13
        int zc = 14 + (lane >> 4) * 3;
#pragma unroll
        for (int q = 0; q < 3; ++q) w->ain[row * 32 + zc + q] = (_Float16)0.0f;
        const float* xr = P.x + (size_t)(g0 + row) * 37 + 13 + 5 * p;
        int cb = (lane >> 4) * 2;
#pragma unroll
        for (int q = 0; q < 2; ++q) w->ain[row * 32 + cb + q] = (_Float16)xr[cb + q];
        const float* hs = P.opp_h + ((size_t)(p * 4) * BATCH + g0) * 10;
#pragma unroll
        for (int q = 0; q < 5; ++q)
          w->ain[row * 32 + 4 + kb + q] = (_Float16)hs[row * 10 + kb + q];
      }
      WAVE_SYNC();

      for (int j = 0; j < 4; ++j) {
        int ci = p * 4 + j;
        cell_gates(sm.tiles + (size_t)(T_OPP + ci * 3) * 512,
                   sm.bias_cell + (10 + ci) * 40, w, lane);
        cell_elem(w, P.opp_c + ((size_t)ci * BATCH + g0) * 10, lane, hout);
        if (j < 3) {
          const float* hs = P.opp_h + ((size_t)(ci + 1) * BATCH + g0) * 10;
#pragma unroll
          for (int q = 0; q < 5; ++q) {
            w->ain[row * 32 + kb + q]      = (_Float16)hout[q];
            w->ain[row * 32 + 10 + kb + q] = (_Float16)hs[row * 10 + kb + q];
          }
        }
#pragma unroll
        for (int q = 0; q < 5; ++q)
          w->cat_opp[row * 64 + 10 * j + kb + q] = (_Float16)hout[q];
        WAVE_SYNC();
      }

      // opp_out_p = tanh(cat_opp(40) @ W1o^T + b1o): 2 chunks x 2 tiles; accumulate mean
      v8f E0, E1;
#pragma unroll
      for (int c = 0; c < 2; ++c) {
        v16h a = load_A(w->cat_opp + c * 32, 64, lane);
        v16h b0 = load_B(sm.tiles + (size_t)(T_W1O + c * 2 + 0) * 512, lane);
        v8f cin0 = (c == 0) ? splat8(sm.b1o[lane & 15]) : E0;
        E0 = wmma16(a, b0, cin0);
        v16h bt1 = load_B(sm.tiles + (size_t)(T_W1O + c * 2 + 1) * 512, lane);
        v8f cin1 = (c == 0) ? splat8(sm.b1o[16 + (lane & 15)]) : E1;
        E1 = wmma16(a, bt1, cin1);
      }
#pragma unroll
      for (int r = 0; r < 8; ++r) { oacc0[r] += ftanh(E0[r]); oacc1[r] += ftanh(E1[r]); }
    }

    // opp_avg -> cat2 cols 50..69
    {
      int col = lane & 15;
      int r0 = (lane >> 4) * 8;
#pragma unroll
      for (int r = 0; r < 8; ++r)
        w->cat2[(r0 + r) * 96 + 50 + col] = (_Float16)(0.2f * oacc0[r]);
      if (col < 4) {
#pragma unroll
        for (int r = 0; r < 8; ++r)
          w->cat2[(r0 + r) * 96 + 66 + col] = (_Float16)(0.2f * oacc1[r]);
      }
    }
    WAVE_SYNC();

    // ---- h2 = tanh(cat2(70) @ W2^T + b2): 3 chunks x 1 tile ----
    v8f F;
#pragma unroll
    for (int c = 0; c < 3; ++c) {
      v16h a = load_A(w->cat2 + c * 32, 96, lane);
      v16h b = load_B(sm.tiles + (size_t)(T_W2 + c) * 512, lane);
      v8f cin = (c == 0) ? splat8(sm.b2[lane & 15]) : F;
      F = wmma16(a, b, cin);
    }
    {
      float* ph = w->h2cm + (lane & 15) * 16 + (lane >> 4) * 8;
      *(float4*)ph       = make_float4(ftanh(F[0]), ftanh(F[1]), ftanh(F[2]), ftanh(F[3]));
      *(float4*)(ph + 4) = make_float4(ftanh(F[4]), ftanh(F[5]), ftanh(F[6]), ftanh(F[7]));
    }
    WAVE_SYNC();

    // ---- out = tanh(h2 @ W3^T + b3) ----
    if (lane < 16) {
      float acc = sm.b3;
#pragma unroll
      for (int j = 0; j < 10; ++j) acc += w->h2cm[j * 16 + lane] * sm.w3[j];
      P.out[g0 + lane] = ftanh(acc);
    }
  }
}

extern "C" void kernel_launch(void* const* d_in, const int* in_sizes, int n_in,
                              void* d_out, int out_size, void* d_ws, size_t ws_size,
                              hipStream_t stream) {
  (void)in_sizes; (void)n_in; (void)out_size; (void)d_ws; (void)ws_size;
  Params P;
  P.x       = (const float*)d_in[0];
  P.gen_h   = (const float*)d_in[1];
  P.gen_c   = (const float*)d_in[2];
  P.opp_h   = (const float*)d_in[3];
  P.opp_c   = (const float*)d_in[4];
  P.W_g0_ih = (const float*)d_in[5];
  P.W_g0_hh = (const float*)d_in[6];
  P.b_g0_ih = (const float*)d_in[7];
  P.b_g0_hh = (const float*)d_in[8];
  P.W_g_ih  = (const float*)d_in[9];
  P.W_g_hh  = (const float*)d_in[10];
  P.b_g_ih  = (const float*)d_in[11];
  P.b_g_hh  = (const float*)d_in[12];
  P.W_o0_ih = (const float*)d_in[13];
  P.W_o0_hh = (const float*)d_in[14];
  P.b_o0_ih = (const float*)d_in[15];
  P.b_o0_hh = (const float*)d_in[16];
  P.W_o_ih  = (const float*)d_in[17];
  P.W_o_hh  = (const float*)d_in[18];
  P.b_o_ih  = (const float*)d_in[19];
  P.b_o_hh  = (const float*)d_in[20];
  P.W1      = (const float*)d_in[21];
  P.b1      = (const float*)d_in[22];
  P.W1o     = (const float*)d_in[23];
  P.b1o     = (const float*)d_in[24];
  P.W2      = (const float*)d_in[25];
  P.b2      = (const float*)d_in[26];
  P.W3      = (const float*)d_in[27];
  P.b3      = (const float*)d_in[28];
  P.out     = (float*)d_out;
  net_kernel<<<dim3(NBLOCKS), dim3(256), 0, stream>>>(P);
}
